// SPE_2508260900995
// MI455X (gfx1250) — matrix operand order repaired and checked by reading, not compiled, not run on hardware
//
#include <hip/hip_runtime.h>
#include <hip/hip_bf16.h>

typedef _Float16 h16;
typedef __attribute__((ext_vector_type(16))) _Float16 v16h;
typedef __attribute__((ext_vector_type(8)))  _Float16 v8h;
typedef __attribute__((ext_vector_type(8)))  float    v8f;
typedef __attribute__((ext_vector_type(4)))  float    v4f;

#define NS  16
#define C   256
#define MID 128
#define HW  4096
#define K9  2304   // 9 * 256

// ---------------- WMMA helpers ----------------

__device__ __forceinline__ v8f wmma_f16(v16h a, v16h b, v8f c) {
  return __builtin_amdgcn_wmma_f32_16x16x32_f16(false, a, false, b, (short)0, c, false, false);
}

// A fragment (16x32 f16): lane row = lane%16; element e -> K = hi*16 + e.
__device__ __forceinline__ v16h a_frag_f16(const h16* __restrict__ base, int hi) {
  const h16* p = base + hi * 16;
  v8h lo = *(const v8h*)(p);
  v8h hh = *(const v8h*)(p + 8);
  v16h r;
#pragma unroll
  for (int i = 0; i < 8; ++i) { r[i] = lo[i]; r[8 + i] = hh[i]; }
  return r;
}

__device__ __forceinline__ v16h a_frag_f32(const float* __restrict__ base, int hi) {
  const float* p = base + hi * 16;
  v4f a0 = *(const v4f*)(p);
  v4f a1 = *(const v4f*)(p + 4);
  v4f b0 = *(const v4f*)(p + 8);
  v4f b1 = *(const v4f*)(p + 12);
  v16h r;
#pragma unroll
  for (int i = 0; i < 4; ++i) {
    r[i]      = (h16)a0[i];
    r[4 + i]  = (h16)a1[i];
    r[8 + i]  = (h16)b0[i];
    r[12 + i] = (h16)b1[i];
  }
  return r;
}

// B fragment (32x16 f16): lane col = lane%16; element e -> K = hi*16 + e (16 contiguous K)
// => one 32-byte contiguous load from NHWC activations.

__device__ __forceinline__ int refl(int v) {  // reflect pad=1 on size 64
  return v < 0 ? -v : (v > 63 ? 126 - v : v);
}

// ---------------- layout prep kernels ----------------

// x: NCHW f32 [16][256][4096] -> xt: NHWC f16 [(s*4096+p)*256 + c]
__global__ __launch_bounds__(256) void transpose_x(const float* __restrict__ x, h16* __restrict__ xt) {
  __shared__ h16 tile[64][65];
  const int b = blockIdx.x;          // 16 * 4 * 64
  const int s = b >> 8;
  const int cb = (b >> 6) & 3;
  const int pb = b & 63;
  const int c0 = cb * 64, p0 = pb * 64;
  const int t = threadIdx.x;
#pragma unroll
  for (int i = 0; i < 16; ++i) {
    const int c = i * 4 + (t >> 6), p = t & 63;
    tile[c][p] = (h16)x[((long)(s * C + c0 + c)) * HW + p0 + p];
  }
  __syncthreads();
#pragma unroll
  for (int i = 0; i < 16; ++i) {
    const int p = i * 4 + (t >> 6), c = t & 63;
    xt[((long)(s * HW) + p0 + p) * C + c0 + c] = tile[c][p];
  }
}

// w[M][256][3][3] f32 -> A[M][9][256] f16 (k = rc*256 + ci)
__global__ __launch_bounds__(256) void repack_w(const float* __restrict__ w, h16* __restrict__ A) {
  const int idx = blockIdx.x * 256 + threadIdx.x;
  const int m = idx / K9;
  const int k = idx - m * K9;
  const int rc = k >> 8, ci = k & 255;
  A[idx] = (h16)w[((long)(m * C + ci)) * 9 + rc];
}

// ---------------- heavy WMMA kernels ----------------
// Each wave: 32 M rows x 64 N pixels, register double-buffered over K.

// conv1: 3x3 reflect, 256->128, lrelu. A1 f16 [128][2304], xt NHWC -> xh_mid NHWC f16 [.][128]
__global__ __launch_bounds__(256) void conv1_kernel(const h16* __restrict__ A1, const h16* __restrict__ xt,
                                                    h16* __restrict__ xh_mid) {
  const int lane = threadIdx.x & 31, wave = threadIdx.x >> 5;
  const int hi = lane >> 4, ln = lane & 15;
  const int s = blockIdx.x >> 5, nb = blockIdx.x & 31;   // 512 blocks
  const int tileM = (wave & 3) * 32;                     // 4 M-tiles cover 128
  const int pBase = nb * 128 + (wave >> 2) * 64;         // 2 pixel groups per block
  const long spBase = (long)s * HW;
  int p[4], py[4], px[4];
  int yc0[4], yc2[4], xc0[4], xc2[4];
#pragma unroll
  for (int j = 0; j < 4; ++j) {
    p[j] = pBase + j * 16 + ln; py[j] = p[j] >> 6; px[j] = p[j] & 63;
    yc0[j] = refl(py[j] - 1); yc2[j] = refl(py[j] + 1);
    xc0[j] = refl(px[j] - 1); xc2[j] = refl(px[j] + 1);
  }
  const h16* aRow0 = A1 + (long)(tileM + ln) * K9;
  const h16* aRow1 = aRow0 + 16 * K9;

  auto calcAddr = [&](int rc, long (&ad)[4]) {
    const int d = rc / 3, e = rc - d * 3;
#pragma unroll
    for (int j = 0; j < 4; ++j) {
      const int y = (d == 0) ? yc0[j] : ((d == 1) ? py[j] : yc2[j]);
      const int x = (e == 0) ? xc0[j] : ((e == 1) ? px[j] : xc2[j]);
      ad[j] = (spBase + y * 64 + x) * C + hi * 16;
    }
  };

  v8f acc0[4] = {}, acc1[4] = {};
  long curA[4], nxtA[4];
  calcAddr(0, curA);
#pragma unroll
  for (int j = 0; j < 4; ++j) nxtA[j] = curA[j];
  v16h a0 = a_frag_f16(aRow0, hi), a1 = a_frag_f16(aRow1, hi);
  v16h bf[4];
#pragma unroll
  for (int j = 0; j < 4; ++j) bf[j] = *(const v16h*)(xt + curA[j]);

  for (int rc = 0; rc < 9; ++rc) {
    const int kR = rc * 256;
    if (rc < 8) calcAddr(rc + 1, nxtA);
#pragma unroll
    for (int kc = 0; kc < 256; kc += 32) {
      v16h na0, na1, nb4[4];
      if (kc < 224) {
        na0 = a_frag_f16(aRow0 + kR + kc + 32, hi);
        na1 = a_frag_f16(aRow1 + kR + kc + 32, hi);
#pragma unroll
        for (int j = 0; j < 4; ++j) nb4[j] = *(const v16h*)(xt + curA[j] + kc + 32);
      } else if (rc < 8) {
        na0 = a_frag_f16(aRow0 + kR + 256, hi);
        na1 = a_frag_f16(aRow1 + kR + 256, hi);
#pragma unroll
        for (int j = 0; j < 4; ++j) nb4[j] = *(const v16h*)(xt + nxtA[j]);
      } else {
        na0 = a0; na1 = a1;
#pragma unroll
        for (int j = 0; j < 4; ++j) nb4[j] = bf[j];
      }
#pragma unroll
      for (int j = 0; j < 4; ++j) {
        acc0[j] = wmma_f16(a0, bf[j], acc0[j]);
        acc1[j] = wmma_f16(a1, bf[j], acc1[j]);
      }
      a0 = na0; a1 = na1;
#pragma unroll
      for (int j = 0; j < 4; ++j) bf[j] = nb4[j];
    }
#pragma unroll
    for (int j = 0; j < 4; ++j) curA[j] = nxtA[j];
  }

  const int m0 = tileM + 8 * hi, m1 = m0 + 16;
#pragma unroll
  for (int j = 0; j < 4; ++j) {
    v8h h0, h1;
#pragma unroll
    for (int r = 0; r < 8; ++r) {
      float v0 = acc0[j][r]; v0 = v0 > 0.f ? v0 : 0.2f * v0; h0[r] = (h16)v0;
      float v1 = acc1[j][r]; v1 = v1 > 0.f ? v1 : 0.2f * v1; h1[r] = (h16)v1;
    }
    h16* o = xh_mid + (spBase + p[j]) * MID;
    *(v8h*)(o + m0) = h0;
    *(v8h*)(o + m1) = h1;
  }
}

// conv2: 1x1, 128->256, lrelu. W f32 [256][128], xm NHWC f16 -> xh32 NCHW f32 + xh16 NHWC f16
__global__ __launch_bounds__(256) void gemm_conv2(const float* __restrict__ W, const h16* __restrict__ xm,
                                                  float* __restrict__ xh32, h16* __restrict__ xh16) {
  const int lane = threadIdx.x & 31, wave = threadIdx.x >> 5;
  const int hi = lane >> 4, ln = lane & 15;
  const int s = blockIdx.x >> 6, nb = blockIdx.x & 63;   // 1024 blocks
  const int tileM = wave * 32;                           // 8 waves cover 256
  const int pBase = nb * 64;
  const long spBase = (long)s * HW;
  const float* aRow0 = W + (long)(tileM + ln) * MID;
  const float* aRow1 = aRow0 + 16 * MID;
  int p[4]; long bBase[4];
#pragma unroll
  for (int j = 0; j < 4; ++j) { p[j] = pBase + j * 16 + ln; bBase[j] = (spBase + p[j]) * MID + hi * 16; }

  v8f acc0[4] = {}, acc1[4] = {};
  v16h a0 = a_frag_f32(aRow0, hi), a1 = a_frag_f32(aRow1, hi);
  v16h bf[4];
#pragma unroll
  for (int j = 0; j < 4; ++j) bf[j] = *(const v16h*)(xm + bBase[j]);
#pragma unroll
  for (int k = 32; k < MID; k += 32) {
    v16h na0 = a_frag_f32(aRow0 + k, hi);
    v16h na1 = a_frag_f32(aRow1 + k, hi);
    v16h nb4[4];
#pragma unroll
    for (int j = 0; j < 4; ++j) nb4[j] = *(const v16h*)(xm + bBase[j] + k);
#pragma unroll
    for (int j = 0; j < 4; ++j) {
      acc0[j] = wmma_f16(a0, bf[j], acc0[j]);
      acc1[j] = wmma_f16(a1, bf[j], acc1[j]);
    }
    a0 = na0; a1 = na1;
#pragma unroll
    for (int j = 0; j < 4; ++j) bf[j] = nb4[j];
  }
#pragma unroll
  for (int j = 0; j < 4; ++j) {
    acc0[j] = wmma_f16(a0, bf[j], acc0[j]);
    acc1[j] = wmma_f16(a1, bf[j], acc1[j]);
  }

  const int m0 = tileM + 8 * hi, m1 = m0 + 16;
#pragma unroll
  for (int j = 0; j < 4; ++j) {
    v8h h0, h1;
#pragma unroll
    for (int r = 0; r < 8; ++r) {
      float v0 = acc0[j][r]; v0 = v0 > 0.f ? v0 : 0.2f * v0;
      float v1 = acc1[j][r]; v1 = v1 > 0.f ? v1 : 0.2f * v1;
      xh32[((long)(s * C + m0 + r)) * HW + p[j]] = v0;
      xh32[((long)(s * C + m1 + r)) * HW + p[j]] = v1;
      h0[r] = (h16)v0; h1[r] = (h16)v1;
    }
    h16* o = xh16 + (spBase + p[j]) * C;
    *(v8h*)(o + m0) = h0;
    *(v8h*)(o + m1) = h1;
  }
}

// be = edge_w @ xh + edge_b. W f32 [256][256], xh16 NHWC -> be NHWC f16
__global__ __launch_bounds__(256) void gemm_edge(const float* __restrict__ W, const float* __restrict__ bias,
                                                 const h16* __restrict__ xh16, h16* __restrict__ be) {
  const int lane = threadIdx.x & 31, wave = threadIdx.x >> 5;
  const int hi = lane >> 4, ln = lane & 15;
  const int s = blockIdx.x >> 6, nb = blockIdx.x & 63;   // 1024 blocks
  const int tileM = wave * 32;
  const int pBase = nb * 64;
  const long spBase = (long)s * HW;
  const float* aRow0 = W + (long)(tileM + ln) * C;
  const float* aRow1 = aRow0 + 16 * C;
  int p[4]; long bBase[4];
#pragma unroll
  for (int j = 0; j < 4; ++j) { p[j] = pBase + j * 16 + ln; bBase[j] = (spBase + p[j]) * C + hi * 16; }

  v8f acc0[4] = {}, acc1[4] = {};
  v16h a0 = a_frag_f32(aRow0, hi), a1 = a_frag_f32(aRow1, hi);
  v16h bf[4];
#pragma unroll
  for (int j = 0; j < 4; ++j) bf[j] = *(const v16h*)(xh16 + bBase[j]);
#pragma unroll
  for (int k = 32; k < C; k += 32) {
    v16h na0 = a_frag_f32(aRow0 + k, hi);
    v16h na1 = a_frag_f32(aRow1 + k, hi);
    v16h nb4[4];
#pragma unroll
    for (int j = 0; j < 4; ++j) nb4[j] = *(const v16h*)(xh16 + bBase[j] + k);
#pragma unroll
    for (int j = 0; j < 4; ++j) {
      acc0[j] = wmma_f16(a0, bf[j], acc0[j]);
      acc1[j] = wmma_f16(a1, bf[j], acc1[j]);
    }
    a0 = na0; a1 = na1;
#pragma unroll
    for (int j = 0; j < 4; ++j) bf[j] = nb4[j];
  }
#pragma unroll
  for (int j = 0; j < 4; ++j) {
    acc0[j] = wmma_f16(a0, bf[j], acc0[j]);
    acc1[j] = wmma_f16(a1, bf[j], acc1[j]);
  }

  const int m0 = tileM + 8 * hi, m1 = m0 + 16;
#pragma unroll
  for (int j = 0; j < 4; ++j) {
    v8h h0, h1;
#pragma unroll
    for (int r = 0; r < 8; ++r) {
      h0[r] = (h16)(acc0[j][r] + bias[m0 + r]);
      h1[r] = (h16)(acc1[j][r] + bias[m1 + r]);
    }
    h16* o = be + (spBase + p[j]) * C;
    *(v8h*)(o + m0) = h0;
    *(v8h*)(o + m1) = h1;
  }
}

// agg = Ai @ be (per sample). Ai f16 [s][256][256], be NHWC -> agg NHWC f16
__global__ __launch_bounds__(256) void gemm_agg(const h16* __restrict__ Ai, const h16* __restrict__ be,
                                                h16* __restrict__ agg) {
  const int lane = threadIdx.x & 31, wave = threadIdx.x >> 5;
  const int hi = lane >> 4, ln = lane & 15;
  const int s = blockIdx.x >> 6, nb = blockIdx.x & 63;   // 1024 blocks
  const int tileM = wave * 32;
  const int pBase = nb * 64;
  const long spBase = (long)s * HW;
  const h16* aRow0 = Ai + ((long)s * C + tileM + ln) * C;
  const h16* aRow1 = aRow0 + 16 * C;
  int p[4]; long bBase[4];
#pragma unroll
  for (int j = 0; j < 4; ++j) { p[j] = pBase + j * 16 + ln; bBase[j] = (spBase + p[j]) * C + hi * 16; }

  v8f acc0[4] = {}, acc1[4] = {};
  v16h a0 = a_frag_f16(aRow0, hi), a1 = a_frag_f16(aRow1, hi);
  v16h bf[4];
#pragma unroll
  for (int j = 0; j < 4; ++j) bf[j] = *(const v16h*)(be + bBase[j]);
#pragma unroll
  for (int k = 32; k < C; k += 32) {
    v16h na0 = a_frag_f16(aRow0 + k, hi);
    v16h na1 = a_frag_f16(aRow1 + k, hi);
    v16h nb4[4];
#pragma unroll
    for (int j = 0; j < 4; ++j) nb4[j] = *(const v16h*)(be + bBase[j] + k);
#pragma unroll
    for (int j = 0; j < 4; ++j) {
      acc0[j] = wmma_f16(a0, bf[j], acc0[j]);
      acc1[j] = wmma_f16(a1, bf[j], acc1[j]);
    }
    a0 = na0; a1 = na1;
#pragma unroll
    for (int j = 0; j < 4; ++j) bf[j] = nb4[j];
  }
#pragma unroll
  for (int j = 0; j < 4; ++j) {
    acc0[j] = wmma_f16(a0, bf[j], acc0[j]);
    acc1[j] = wmma_f16(a1, bf[j], acc1[j]);
  }

  const int m0 = tileM + 8 * hi, m1 = m0 + 16;
#pragma unroll
  for (int j = 0; j < 4; ++j) {
    v8h h0, h1;
#pragma unroll
    for (int r = 0; r < 8; ++r) { h0[r] = (h16)acc0[j][r]; h1[r] = (h16)acc1[j][r]; }
    h16* o = agg + (spBase + p[j]) * C;
    *(v8h*)(o + m0) = h0;
    *(v8h*)(o + m1) = h1;
  }
}

// out conv: 3x3 reflect on agg (256->256) + out_b + residual(xh32) -> y_out f32 NCHW
__global__ __launch_bounds__(256) void conv_out_kernel(const h16* __restrict__ A6, const h16* __restrict__ agg,
                                                       const float* __restrict__ out_b,
                                                       const float* __restrict__ xh32,
                                                       float* __restrict__ out) {
  const int lane = threadIdx.x & 31, wave = threadIdx.x >> 5;
  const int hi = lane >> 4, ln = lane & 15;
  const int s = blockIdx.x >> 6, nb = blockIdx.x & 63;   // 1024 blocks
  const int tileM = wave * 32;
  const int pBase = nb * 64;
  const long spBase = (long)s * HW;
  int p[4], py[4], px[4];
  int yc0[4], yc2[4], xc0[4], xc2[4];
#pragma unroll
  for (int j = 0; j < 4; ++j) {
    p[j] = pBase + j * 16 + ln; py[j] = p[j] >> 6; px[j] = p[j] & 63;
    yc0[j] = refl(py[j] - 1); yc2[j] = refl(py[j] + 1);
    xc0[j] = refl(px[j] - 1); xc2[j] = refl(px[j] + 1);
  }
  const h16* aRow0 = A6 + (long)(tileM + ln) * K9;
  const h16* aRow1 = aRow0 + 16 * K9;

  auto calcAddr = [&](int rc, long (&ad)[4]) {
    const int d = rc / 3, e = rc - d * 3;
#pragma unroll
    for (int j = 0; j < 4; ++j) {
      const int y = (d == 0) ? yc0[j] : ((d == 1) ? py[j] : yc2[j]);
      const int x = (e == 0) ? xc0[j] : ((e == 1) ? px[j] : xc2[j]);
      ad[j] = (spBase + y * 64 + x) * C + hi * 16;
    }
  };

  v8f acc0[4] = {}, acc1[4] = {};
  long curA[4], nxtA[4];
  calcAddr(0, curA);
#pragma unroll
  for (int j = 0; j < 4; ++j) nxtA[j] = curA[j];
  v16h a0 = a_frag_f16(aRow0, hi), a1 = a_frag_f16(aRow1, hi);
  v16h bf[4];
#pragma unroll
  for (int j = 0; j < 4; ++j) bf[j] = *(const v16h*)(agg + curA[j]);

  for (int rc = 0; rc < 9; ++rc) {
    const int kR = rc * 256;
    if (rc < 8) calcAddr(rc + 1, nxtA);
#pragma unroll
    for (int kc = 0; kc < 256; kc += 32) {
      v16h na0, na1, nb4[4];
      if (kc < 224) {
        na0 = a_frag_f16(aRow0 + kR + kc + 32, hi);
        na1 = a_frag_f16(aRow1 + kR + kc + 32, hi);
#pragma unroll
        for (int j = 0; j < 4; ++j) nb4[j] = *(const v16h*)(agg + curA[j] + kc + 32);
      } else if (rc < 8) {
        na0 = a_frag_f16(aRow0 + kR + 256, hi);
        na1 = a_frag_f16(aRow1 + kR + 256, hi);
#pragma unroll
        for (int j = 0; j < 4; ++j) nb4[j] = *(const v16h*)(agg + nxtA[j]);
      } else {
        na0 = a0; na1 = a1;
#pragma unroll
        for (int j = 0; j < 4; ++j) nb4[j] = bf[j];
      }
#pragma unroll
      for (int j = 0; j < 4; ++j) {
        acc0[j] = wmma_f16(a0, bf[j], acc0[j]);
        acc1[j] = wmma_f16(a1, bf[j], acc1[j]);
      }
      a0 = na0; a1 = na1;
#pragma unroll
      for (int j = 0; j < 4; ++j) bf[j] = nb4[j];
    }
#pragma unroll
    for (int j = 0; j < 4; ++j) curA[j] = nxtA[j];
  }

  const int m0 = tileM + 8 * hi, m1 = m0 + 16;
#pragma unroll
  for (int j = 0; j < 4; ++j) {
#pragma unroll
    for (int r = 0; r < 8; ++r) {
      const long o0 = ((long)(s * C + m0 + r)) * HW + p[j];
      const long o1 = ((long)(s * C + m1 + r)) * HW + p[j];
      out[o0] = acc0[j][r] + out_b[m0 + r] + xh32[o0];
      out[o1] = acc1[j][r] + out_b[m1 + r] + xh32[o1];
    }
  }
}

// ---------------- small kernels ----------------

__global__ __launch_bounds__(256) void mean_kernel(const float* __restrict__ xh32, float* __restrict__ mean) {
  __shared__ float red[256];
  const long base = (long)blockIdx.x * HW;   // blockIdx = s*256 + c (NCHW)
  float sum = 0.f;
  for (int i = threadIdx.x; i < HW; i += 256) sum += xh32[base + i];
  red[threadIdx.x] = sum; __syncthreads();
  for (int o = 128; o > 0; o >>= 1) { if (threadIdx.x < o) red[threadIdx.x] += red[threadIdx.x + o]; __syncthreads(); }
  if (threadIdx.x == 0) mean[blockIdx.x] = red[0] * (1.0f / 4096.0f);
}

// wi = node_w@mean + node_b ; xi = edge_w@mean + edge_b ; u = (mc_w+mf_w)@wi  (per sample)
__global__ __launch_bounds__(256) void vec_kernel(const float* __restrict__ mean,
    const float* __restrict__ node_w, const float* __restrict__ node_b,
    const float* __restrict__ edge_w, const float* __restrict__ edge_b,
    const float* __restrict__ mc_w,  const float* __restrict__ mf_w,
    float* __restrict__ wi, float* __restrict__ xi, float* __restrict__ u) {
  __shared__ float msh[C];
  __shared__ float wsh[C];
  const int s = blockIdx.x, i = threadIdx.x;
  msh[i] = mean[s * C + i];
  __syncthreads();
  float aw = node_b[i], ax = edge_b[i];
  for (int c2 = 0; c2 < C; ++c2) {
    aw += node_w[i * C + c2] * msh[c2];
    ax += edge_w[i * C + c2] * msh[c2];
  }
  wi[s * C + i] = aw; xi[s * C + i] = ax; wsh[i] = aw;
  __syncthreads();
  float uu = 0.f;
  for (int c2 = 0; c2 < C; ++c2) uu += (mc_w[i * C + c2] + mf_w[i * C + c2]) * wsh[c2];
  u[s * C + i] = uu;
}

// Ti = u_i*xi_j + bm_i ; mask = sigmoid ; softmax_j(wi_i*xi_j*mask/64) ; Ai = y * (mask!=0)
__global__ __launch_bounds__(256) void mask_kernel(const float* __restrict__ wi, const float* __restrict__ xi,
    const float* __restrict__ u, const float* __restrict__ mc_b, const float* __restrict__ mf_b,
    float* __restrict__ AiOut, h16* __restrict__ Ai16) {
  __shared__ float red[C];
  const int s = blockIdx.x >> 8, i = blockIdx.x & 255, j = threadIdx.x;  // 4096 blocks
  const float w_i = wi[s * C + i];
  const float u_i = u[s * C + i];
  const float bm  = mc_b[i] + mf_b[i];
  const float x_j = xi[s * C + j];
  const float ti  = u_i * x_j + bm;
  const float mk  = 1.f / (1.f + __expf(-ti));
  const float t   = w_i * x_j * mk * (1.0f / 64.0f);   // scale = hw^-0.5
  red[j] = t; __syncthreads();
  for (int o = 128; o > 0; o >>= 1) { if (j < o) red[j] = fmaxf(red[j], red[j + o]); __syncthreads(); }
  const float mx = red[0];
  __syncthreads();
  const float e = __expf(t - mx);
  red[j] = e; __syncthreads();
  for (int o = 128; o > 0; o >>= 1) { if (j < o) red[j] += red[j + o]; __syncthreads(); }
  const float y = e / red[0];
  const float a = (mk != 0.0f) ? y : 0.0f;
  const long o = ((long)(s * C + i) << 8) + j;
  AiOut[o] = a;
  Ai16[o] = (h16)a;
}

// ---------------- launch ----------------

extern "C" void kernel_launch(void* const* d_in, const int* in_sizes, int n_in,
                              void* d_out, int out_size, void* d_ws, size_t ws_size,
                              hipStream_t stream) {
  (void)in_sizes; (void)n_in; (void)out_size; (void)ws_size;
  const float* x       = (const float*)d_in[0];
  const float* conv1_w = (const float*)d_in[1];
  const float* conv2_w = (const float*)d_in[2];
  const float* node_w  = (const float*)d_in[3];
  const float* node_b  = (const float*)d_in[4];
  const float* edge_w  = (const float*)d_in[5];
  const float* edge_b  = (const float*)d_in[6];
  const float* mc_w    = (const float*)d_in[7];
  const float* mc_b    = (const float*)d_in[8];
  const float* mf_w    = (const float*)d_in[9];
  const float* mf_b    = (const float*)d_in[10];
  const float* out_w   = (const float*)d_in[11];
  const float* out_b   = (const float*)d_in[12];

  char* ws = (char*)d_ws;
  size_t off = 0;
  auto take = [&](size_t bytes) { char* p = ws + off; off += (bytes + 255) & ~(size_t)255; return p; };
  h16*   xt     = (h16*)take((size_t)NS * HW * C * 2);    // reused as `be`
  h16*   A1     = (h16*)take((size_t)MID * K9 * 2);
  h16*   A6     = (h16*)take((size_t)C * K9 * 2);
  h16*   xh_mid = (h16*)take((size_t)NS * HW * MID * 2);
  float* xh32   = (float*)take((size_t)NS * C * HW * 4);
  h16*   xh16   = (h16*)take((size_t)NS * HW * C * 2);    // reused as `agg`
  float* mean   = (float*)take((size_t)NS * C * 4);
  float* wi     = (float*)take((size_t)NS * C * 4);
  float* xi     = (float*)take((size_t)NS * C * 4);
  float* u      = (float*)take((size_t)NS * C * 4);
  h16*   Ai16   = (h16*)take((size_t)NS * C * C * 2);
  h16*   be     = xt;     // xt dead after conv1
  h16*   agg    = xh16;   // xh16 dead after gemm_edge

  float* Ai_out = (float*)d_out;
  float* y_out  = (float*)d_out + (size_t)NS * C * C;

  transpose_x    <<<4096, 256, 0, stream>>>(x, xt);
  repack_w       <<<1152, 256, 0, stream>>>(conv1_w, A1);
  repack_w       <<<2304, 256, 0, stream>>>(out_w, A6);
  conv1_kernel   <<<512,  256, 0, stream>>>(A1, xt, xh_mid);
  gemm_conv2     <<<1024, 256, 0, stream>>>(conv2_w, xh_mid, xh32, xh16);
  mean_kernel    <<<4096, 256, 0, stream>>>(xh32, mean);
  vec_kernel     <<<16,   256, 0, stream>>>(mean, node_w, node_b, edge_w, edge_b, mc_w, mf_w, wi, xi, u);
  mask_kernel    <<<4096, 256, 0, stream>>>(wi, xi, u, mc_b, mf_b, Ai_out, Ai16);
  gemm_edge      <<<1024, 256, 0, stream>>>(edge_w, edge_b, xh16, be);
  gemm_agg       <<<1024, 256, 0, stream>>>(Ai16, be, agg);
  conv_out_kernel<<<1024, 256, 0, stream>>>(A6, agg, out_b, xh32, y_out);
}
